// PointerGeneratorNetwork_22402549416698
// MI455X (gfx1250) — compile-verified, matrix operand order checked by reading, not run on hardware
//
#include <hip/hip_runtime.h>
#include <hip/hip_bf16.h>
#include <math.h>

#define V_    32000
#define EMB_  128
#define EH_   256
#define DH_   256
#define AD_   256
#define BN_   512
#define B_    16
#define TIN_  400
#define TTGT_ 100
#define UNK_   0
#define START_ 1
#define END_   2
#define PAD_   3

typedef __attribute__((ext_vector_type(16))) __bf16 v16bf;
typedef __attribute__((ext_vector_type(8)))  __bf16 v8bf;
typedef __attribute__((ext_vector_type(8)))  float  v8f;

__device__ __forceinline__ v8f wmma_bf16(v16bf a, v16bf b, v8f c) {
  return __builtin_amdgcn_wmma_f32_16x16x32_bf16(false, a, false, b, (short)0, c, false, false);
}

// WMMA 16-element operand from an fp32 row (converted): p = row + kb + 8*hi.
// Elements 0..7 <- p[0..7], 8..15 <- p[16..23]  (ISA 7.12.2 16-bit layout).
__device__ __forceinline__ v16bf ldrow_f32(const float* __restrict__ p) {
  const float4* q = (const float4*)p;
  float4 a0 = q[0], a1 = q[1], b0 = q[4], b1 = q[5];
  v16bf v;
  v[0]  = (__bf16)a0.x; v[1]  = (__bf16)a0.y; v[2]  = (__bf16)a0.z; v[3]  = (__bf16)a0.w;
  v[4]  = (__bf16)a1.x; v[5]  = (__bf16)a1.y; v[6]  = (__bf16)a1.z; v[7]  = (__bf16)a1.w;
  v[8]  = (__bf16)b0.x; v[9]  = (__bf16)b0.y; v[10] = (__bf16)b0.z; v[11] = (__bf16)b0.w;
  v[12] = (__bf16)b1.x; v[13] = (__bf16)b1.y; v[14] = (__bf16)b1.z; v[15] = (__bf16)b1.w;
  return v;
}

// WMMA 16-element operand from a pre-converted bf16 row: two 16B loads, no cvt.
__device__ __forceinline__ v16bf ldrow_bf(const __bf16* p) {
  v8bf lo = *(const v8bf*)p;
  v8bf hi = *(const v8bf*)(p + 16);
  v16bf v;
#pragma unroll
  for (int i = 0; i < 8; ++i) { v[i] = lo[i]; v[8 + i] = hi[i]; }
  return v;
}

__device__ __forceinline__ float sigf(float x) { return 1.f / (1.f + expf(-x)); }

__device__ __forceinline__ float wred_sum(float v) {
#pragma unroll
  for (int o = 16; o; o >>= 1) v += __shfl_xor(v, o, 32);
  return v;
}
__device__ __forceinline__ float wred_max(float v) {
#pragma unroll
  for (int o = 16; o; o >>= 1) v = fmaxf(v, __shfl_xor(v, o, 32));
  return v;
}

// ---------------------------------------------------------------------------
// GEMM (f32 A, bf16 W): C = act(A @ W^T + bias), M,N multiples of 16, K%32==0.
// One wave per 16x16 tile. Optionally also emits a bf16 copy of C.
// ---------------------------------------------------------------------------
__global__ void k_gemm_fb(const float* __restrict__ A, int lda,
                          const __bf16* __restrict__ W, int ldw,
                          const float* __restrict__ bias,
                          float* __restrict__ C, __bf16* __restrict__ Cbf, int ldc,
                          int K, int relu) {
  int lane  = threadIdx.x & 31;
  int nbase = blockIdx.x * 16;
  int mbase = blockIdx.y * 16;
  int r = lane & 15;
  int off8 = ((lane >> 4) & 1) << 3;
  const float*  arow = A + (size_t)(mbase + r) * lda + off8;
  const __bf16* wrow = W + (size_t)(nbase + r) * ldw + off8;
  v8f acc = {};
  for (int kb = 0; kb < K; kb += 32) {
    v16bf a = ldrow_f32(arow + kb);
    v16bf b = ldrow_bf(wrow + kb);
    acc = wmma_bf16(a, b, acc);
  }
  int n  = nbase + (lane & 15);
  int m0 = mbase + ((lane >> 4) << 3);
  float bv = bias ? bias[n] : 0.f;
#pragma unroll
  for (int rr = 0; rr < 8; ++rr) {
    float v = acc[rr] + bv;
    if (relu) v = fmaxf(v, 0.f);
    C[(size_t)(m0 + rr) * ldc + n] = v;
    if (Cbf) Cbf[(size_t)(m0 + rr) * ldc + n] = (__bf16)v;
  }
}

// ---------------------------------------------------------------------------
// GEMM (bf16 A, bf16 W), TWO N-tiles per wave: A operand reused for 2 WMMAs.
// Used for the hot vocab projection (N ragged: rows clamped, stores guarded).
// ---------------------------------------------------------------------------
__global__ void k_gemm_bb2(const __bf16* __restrict__ A, int lda,
                           const __bf16* __restrict__ W, int ldw,
                           const float* __restrict__ bias,
                           float* __restrict__ C, int ldc,
                           int N, int K) {
  int lane  = threadIdx.x & 31;
  int nbase = blockIdx.x * 32;
  int mbase = blockIdx.y * 16;
  int r = lane & 15;
  int off8 = ((lane >> 4) & 1) << 3;
  int bn0 = nbase + r;      if (bn0 >= N) bn0 = N - 1;
  int bn1 = nbase + 16 + r; if (bn1 >= N) bn1 = N - 1;
  const __bf16* arow  = A + (size_t)(mbase + r) * lda + off8;
  const __bf16* wrow0 = W + (size_t)bn0 * ldw + off8;
  const __bf16* wrow1 = W + (size_t)bn1 * ldw + off8;
  v8f acc0 = {}, acc1 = {};
  for (int kb = 0; kb < K; kb += 32) {
    v16bf a = ldrow_bf(arow + kb);
    acc0 = wmma_bf16(a, ldrow_bf(wrow0 + kb), acc0);
    acc1 = wmma_bf16(a, ldrow_bf(wrow1 + kb), acc1);
  }
  int m0 = mbase + ((lane >> 4) << 3);
  int n0 = nbase + (lane & 15);
  int n1 = n0 + 16;
  if (n0 < N) {
    float bv = bias ? bias[n0] : 0.f;
#pragma unroll
    for (int rr = 0; rr < 8; ++rr) C[(size_t)(m0 + rr) * ldc + n0] = acc0[rr] + bv;
  }
  if (n1 < N) {
    float bv = bias ? bias[n1] : 0.f;
#pragma unroll
    for (int rr = 0; rr < 8; ++rr) C[(size_t)(m0 + rr) * ldc + n1] = acc1[rr] + bv;
  }
}

// ---------------------------------------------------------------------------
// Persistent LSTM over time, one block per direction, blockDim=512 (16 waves).
// X bf16 time-major [T,16,KX]; weights bf16; h kept bf16 in LDS, c f32 in LDS.
// Wave w owns hidden tile j in [16w,16w+16) and computes all 4 gate tiles.
// Next timestep's X row is prefetched while the current step computes.
// ---------------------------------------------------------------------------
__global__ void k_lstm(const __bf16* __restrict__ X, int KX,
                       const __bf16* __restrict__ Wih_all,  // [2,1024,KX]
                       const __bf16* __restrict__ Whh_all,  // [2,1024,256]
                       const float* __restrict__ b_all,     // [2,1024]
                       __bf16* __restrict__ hseqb,          // [T,16,512] or null
                       float* __restrict__ hseqf,           // [T,16,512] or null
                       float* __restrict__ hT,              // [16,512]
                       float* __restrict__ cT,              // [16,512]
                       int T) {
  int dir = blockIdx.x;
  const __bf16* Wih = Wih_all + (size_t)dir * 1024 * KX;
  const __bf16* Whh = Whh_all + (size_t)dir * 1024 * 256;
  const float* bv = b_all + dir * 1024;
  __shared__ __align__(16) __bf16 shb[4096];
  __shared__ __align__(16) float  sc[4096];
  int tid = threadIdx.x;
  for (int i = tid; i < 4096; i += blockDim.x) { shb[i] = (__bf16)0.f; sc[i] = 0.f; }
  __syncthreads();
  int lane = tid & 31, wave = tid >> 5;
  int r = lane & 15, hi = (lane >> 4) & 1;
  int off8 = hi << 3;
  for (int step = 0; step < T; ++step) {
    int t = dir ? (T - 1 - step) : step;
    // prefetch next step's input row into cache while this step computes
    if (step + 1 < T) {
      int tn = dir ? (T - 2 - step) : (step + 1);
      __builtin_prefetch(X + ((size_t)tn * 16 + r) * KX + off8, 0, 0);
    }
    v8f acc[4] = {};
    const __bf16* xrow = X + ((size_t)t * 16 + r) * KX + off8;
    for (int kb = 0; kb < KX; kb += 32) {
      v16bf a = ldrow_bf(xrow + kb);
#pragma unroll
      for (int g = 0; g < 4; ++g) {
        v16bf bb = ldrow_bf(Wih + (size_t)(g * 256 + wave * 16 + r) * KX + off8 + kb);
        acc[g] = wmma_bf16(a, bb, acc[g]);
      }
    }
    for (int kb = 0; kb < 256; kb += 32) {
      v16bf a = ldrow_bf(&shb[r * 256 + off8 + kb]);
#pragma unroll
      for (int g = 0; g < 4; ++g) {
        v16bf bb = ldrow_bf(Whh + (size_t)(g * 256 + wave * 16 + r) * 256 + off8 + kb);
        acc[g] = wmma_bf16(a, bb, acc[g]);
      }
    }
    __syncthreads();  // reads of shb complete
#pragma unroll
    for (int rr = 0; rr < 8; ++rr) {
      int m = rr + (hi << 3);
      int j = wave * 16 + r;
      float gi = acc[0][rr] + bv[j];
      float gf = acc[1][rr] + bv[256 + j];
      float gg = acc[2][rr] + bv[512 + j];
      float go = acc[3][rr] + bv[768 + j];
      float c = sigf(gf) * sc[m * 256 + j] + sigf(gi) * tanhf(gg);
      float h = sigf(go) * tanhf(c);
      sc[m * 256 + j] = c;
      shb[m * 256 + j] = (__bf16)h;
      size_t oidx = ((size_t)t * 16 + m) * 512 + dir * 256 + j;
      if (hseqb) hseqb[oidx] = (__bf16)h;
      if (hseqf) hseqf[oidx] = h;
    }
    __syncthreads();  // new h visible before next GEMM
  }
#pragma unroll
  for (int rr = 0; rr < 8; ++rr) {
    int m = rr + (hi << 3);
    int j = wave * 16 + r;
    hT[m * 512 + dir * 256 + j] = (float)shb[m * 256 + j];
    cT[m * 512 + dir * 256 + j] = sc[m * 256 + j];
  }
}

// ---------------------------------------------------------------------------
// Decoder step: 2 LSTM cells + attention projection + switch partials.
// One block, 512 threads (16 waves). States bf16 in LDS, c f32 in global.
// ---------------------------------------------------------------------------
__global__ void k_dec_step(int t,
    const __bf16* __restrict__ edec,  // [TTGT,16,128] bf16
    const __bf16* __restrict__ Wih0, const __bf16* __restrict__ Whh0, const float* __restrict__ b0,
    const __bf16* __restrict__ Wih1, const __bf16* __restrict__ Whh1, const float* __restrict__ b1,
    const __bf16* __restrict__ Wdatt,    // [256,256] bf16
    const float* __restrict__ w_dec_sw,  // [256]
    const float* __restrict__ w_emb_sw,  // [128]
    const float* __restrict__ b_ctx_sw,  // [1]
    float* __restrict__ dh,   // [2,16,256]
    float* __restrict__ dc,   // [2,16,256]
    float* __restrict__ hc,   // [16,768]: h1 -> cols 0..255
    float* __restrict__ dproj,// [16,256]
    float* __restrict__ hsw)  // [16]
{
  __shared__ __align__(16) __bf16 s_a[4096];    // dh0; reused later for h1
  __shared__ __align__(16) __bf16 s_dh1[4096];
  __shared__ __align__(16) __bf16 s_h0[4096];
  int tid = threadIdx.x, lane = tid & 31, wave = tid >> 5;
  int r = lane & 15, hi = (lane >> 4) & 1;
  int off8 = hi << 3;
  // prefetch next step's embedding row
  if (t + 1 < TTGT_) __builtin_prefetch(edec + ((size_t)(t + 1) * 16 + r) * 128 + off8, 0, 0);
  for (int i = tid; i < 4096; i += blockDim.x) {
    s_a[i] = (__bf16)dh[i];
    s_dh1[i] = (__bf16)dh[4096 + i];
  }
  __syncthreads();
  // ---- cell 0: gates = e_t @ Wih0^T + dh0 @ Whh0^T + b0
  {
    v8f acc[4] = {};
    const __bf16* erow = edec + ((size_t)t * 16 + r) * 128 + off8;
    for (int kb = 0; kb < 128; kb += 32) {
      v16bf a = ldrow_bf(erow + kb);
#pragma unroll
      for (int g = 0; g < 4; ++g) {
        v16bf bb = ldrow_bf(Wih0 + (size_t)(g * 256 + wave * 16 + r) * 128 + off8 + kb);
        acc[g] = wmma_bf16(a, bb, acc[g]);
      }
    }
    for (int kb = 0; kb < 256; kb += 32) {
      v16bf a = ldrow_bf(&s_a[r * 256 + off8 + kb]);
#pragma unroll
      for (int g = 0; g < 4; ++g) {
        v16bf bb = ldrow_bf(Whh0 + (size_t)(g * 256 + wave * 16 + r) * 256 + off8 + kb);
        acc[g] = wmma_bf16(a, bb, acc[g]);
      }
    }
    __syncthreads();
#pragma unroll
    for (int rr = 0; rr < 8; ++rr) {
      int m = rr + (hi << 3);
      int j = wave * 16 + r;
      float gi = acc[0][rr] + b0[j];
      float gf = acc[1][rr] + b0[256 + j];
      float gg = acc[2][rr] + b0[512 + j];
      float go = acc[3][rr] + b0[768 + j];
      float c = sigf(gf) * dc[m * 256 + j] + sigf(gi) * tanhf(gg);
      float h = sigf(go) * tanhf(c);
      dc[m * 256 + j] = c; dh[m * 256 + j] = h; s_h0[m * 256 + j] = (__bf16)h;
    }
    __syncthreads();
  }
  // ---- cell 1: gates = h0 @ Wih1^T + dh1 @ Whh1^T + b1
  {
    v8f acc[4] = {};
    for (int kb = 0; kb < 256; kb += 32) {
      v16bf a = ldrow_bf(&s_h0[r * 256 + off8 + kb]);
#pragma unroll
      for (int g = 0; g < 4; ++g) {
        v16bf bb = ldrow_bf(Wih1 + (size_t)(g * 256 + wave * 16 + r) * 256 + off8 + kb);
        acc[g] = wmma_bf16(a, bb, acc[g]);
      }
    }
    for (int kb = 0; kb < 256; kb += 32) {
      v16bf a = ldrow_bf(&s_dh1[r * 256 + off8 + kb]);
#pragma unroll
      for (int g = 0; g < 4; ++g) {
        v16bf bb = ldrow_bf(Whh1 + (size_t)(g * 256 + wave * 16 + r) * 256 + off8 + kb);
        acc[g] = wmma_bf16(a, bb, acc[g]);
      }
    }
    __syncthreads();
#pragma unroll
    for (int rr = 0; rr < 8; ++rr) {
      int m = rr + (hi << 3);
      int j = wave * 16 + r;
      float gi = acc[0][rr] + b1[j];
      float gf = acc[1][rr] + b1[256 + j];
      float gg = acc[2][rr] + b1[512 + j];
      float go = acc[3][rr] + b1[768 + j];
      float c = sigf(gf) * dc[4096 + m * 256 + j] + sigf(gi) * tanhf(gg);
      float h = sigf(go) * tanhf(c);
      dc[4096 + m * 256 + j] = c; dh[4096 + m * 256 + j] = h;
      s_a[m * 256 + j] = (__bf16)h;   // h1
      hc[m * 768 + j] = h;
    }
    __syncthreads();
  }
  // ---- dproj = h1 @ Wdatt^T : N=256, wave w owns tile w. K=256.
  {
    v8f acc = {};
    for (int kb = 0; kb < 256; kb += 32) {
      v16bf a  = ldrow_bf(&s_a[r * 256 + off8 + kb]);
      v16bf bb = ldrow_bf(Wdatt + (size_t)(wave * 16 + r) * 256 + off8 + kb);
      acc = wmma_bf16(a, bb, acc);
    }
#pragma unroll
    for (int rr = 0; rr < 8; ++rr) {
      int m = rr + (hi << 3);
      int n = wave * 16 + (lane & 15);
      dproj[m * 256 + n] = acc[rr];
    }
  }
  __syncthreads();
  if (tid < 16) {
    float s = b_ctx_sw[0];
    for (int k = 0; k < 256; ++k) s += (float)s_a[tid * 256 + k] * w_dec_sw[k];
    for (int k = 0; k < 128; ++k) s += (float)edec[((size_t)t * 16 + tid) * 128 + k] * w_emb_sw[k];
    hsw[tid] = s;
  }
}

// ---------------------------------------------------------------------------
// Attention scores: one wave per (b,t).
// ---------------------------------------------------------------------------
__global__ void k_scores(const float* __restrict__ enc_attn,  // [T,16,256]
                         const float* __restrict__ dproj,     // [16,256]
                         const float* __restrict__ coverage,  // [16,400]
                         const float* __restrict__ w_cov,
                         const float* __restrict__ v_attn,
                         const int* __restrict__ input_ids,   // [16,400]
                         float* __restrict__ scores) {        // [16,400]
  int blk = blockIdx.x;
  int b = blk / TIN_, t = blk % TIN_;
  int lane = threadIdx.x;
  float cov = coverage[b * TIN_ + t];
  float s = 0.f;
  for (int d = lane; d < 256; d += 32) {
    float f = tanhf(enc_attn[((size_t)t * 16 + b) * 256 + d] + dproj[b * 256 + d] + cov * w_cov[d]);
    s += f * v_attn[d];
  }
  s = wred_sum(s);
  if (lane == 0) {
    bool pad = input_ids[b * TIN_ + t] == PAD_;
    scores[b * TIN_ + t] = pad ? -INFINITY : s;
  }
}

// ---------------------------------------------------------------------------
// Per-batch softmax over T_IN, context, p_copy, coverage min/update.
// grid = 16, blockDim = 512 (16 waves).
// ---------------------------------------------------------------------------
__global__ void k_attn_ctx(int t,
    const float* __restrict__ scores,
    const float* __restrict__ enc_states,  // [T,16,512]
    const int* __restrict__ input_ids,
    const int* __restrict__ target_ids,    // [16,100]
    float* __restrict__ coverage,          // [16,400]
    float* __restrict__ hc,                // ctx -> cols 256..767
    float* __restrict__ p_copy,
    float* __restrict__ covstep) {
  int b = blockIdx.x;
  int tid = threadIdx.x, lane = tid & 31, wave = tid >> 5;
  __shared__ float s_attn[TIN_];
  __shared__ float red[16];
  __shared__ float sval;
  float sc = (tid < TIN_) ? scores[b * TIN_ + tid] : -INFINITY;
  float m = wred_max(sc);
  if (lane == 0) red[wave] = m;
  __syncthreads();
  if (tid == 0) { float mm = red[0]; for (int w = 1; w < 16; ++w) mm = fmaxf(mm, red[w]); sval = mm; }
  __syncthreads();
  float bmax = sval;
  float e = (tid < TIN_) ? expf(sc - bmax) : 0.f;
  float se = wred_sum(e);
  if (lane == 0) red[wave] = se;
  __syncthreads();
  if (tid == 0) { float ss = 0.f; for (int w = 0; w < 16; ++w) ss += red[w]; sval = ss; }
  __syncthreads();
  float attn = e / sval;
  if (tid < TIN_) s_attn[tid] = attn;
  int tgt = target_ids[b * TTGT_ + t];
  float pc = (tid < TIN_ && input_ids[b * TIN_ + tid] == tgt) ? attn : 0.f;
  float covold = (tid < TIN_) ? coverage[b * TIN_ + tid] : 0.f;
  float cmin = fminf(attn, covold);
  float pcw = wred_sum(pc);
  if (lane == 0) red[wave] = pcw;
  __syncthreads();
  if (tid == 0) { float ss = 0.f; for (int w = 0; w < 16; ++w) ss += red[w]; p_copy[b] = ss; }
  __syncthreads();
  float cmw = wred_sum(cmin);
  if (lane == 0) red[wave] = cmw;
  __syncthreads();
  if (tid == 0) { float ss = 0.f; for (int w = 0; w < 16; ++w) ss += red[w]; covstep[b] = ss; }
  if (tid < TIN_) coverage[b * TIN_ + tid] = covold + attn;
  __syncthreads();
  // ctx[d] = sum_t attn[t] * enc_states[t,b,d]
  if (tid < 512) {
    float acc = 0.f;
    for (int tt = 0; tt < TIN_; ++tt) acc += s_attn[tt] * enc_states[((size_t)tt * 16 + b) * 512 + tid];
    hc[b * 768 + 256 + tid] = acc;
  }
}

// ---------------------------------------------------------------------------
// Per-batch logsumexp stats over the 31998 logits. grid=16, blockDim=1024.
// ---------------------------------------------------------------------------
__global__ void k_softmax_stats(const float* __restrict__ logits,  // [16, 32000]
                                float* __restrict__ mx, float* __restrict__ sm) {
  int b = blockIdx.x, tid = threadIdx.x, lane = tid & 31, wave = tid >> 5;
  __shared__ float red[32];
  __shared__ float sval;
  float m = -INFINITY;
  for (int n = tid; n < V_ - END_; n += 1024) m = fmaxf(m, logits[(size_t)b * 32000 + n]);
  m = wred_max(m);
  if (lane == 0) red[wave] = m;
  __syncthreads();
  if (tid == 0) { float mm = red[0]; for (int w = 1; w < 32; ++w) mm = fmaxf(mm, red[w]); sval = mm; }
  __syncthreads();
  float bm = sval;
  float s = 0.f;
  for (int n = tid; n < V_ - END_; n += 1024) s += expf(logits[(size_t)b * 32000 + n] - bm);
  s = wred_sum(s);
  if (lane == 0) red[wave] = s;
  __syncthreads();
  if (tid == 0) { float ss = 0.f; for (int w = 0; w < 32; ++w) ss += red[w]; mx[b] = bm; sm[b] = ss; }
}

// ---------------------------------------------------------------------------
// Final per-step mix: p_gen, p_vocab, p_copy -> nll/cov accumulation.
// ---------------------------------------------------------------------------
__global__ void k_finalize(int t,
    const float* __restrict__ logits,
    const float* __restrict__ mx, const float* __restrict__ sm,
    const float* __restrict__ hc,        // ctx at cols 256..767
    const float* __restrict__ w_ctx_sw,  // [512]
    const float* __restrict__ hsw,
    const float* __restrict__ p_copy,
    const float* __restrict__ covstep,
    const int* __restrict__ target_ids,
    const int* __restrict__ tlen,
    float* __restrict__ nll, float* __restrict__ cov) {
  int b = blockIdx.x, lane = threadIdx.x;
  float s = 0.f;
  for (int d = lane; d < 512; d += 32) s += hc[b * 768 + 256 + d] * w_ctx_sw[d];
  s = wred_sum(s);
  if (lane == 0) {
    float pg = sigf(s + hsw[b]);
    int tgt = target_ids[b * TTGT_ + t];
    float pv = 0.f;
    if (tgt >= END_) pv = expf(logits[(size_t)b * 32000 + (tgt - END_)] - mx[b]) / sm[b];
    float p = pg * pv + (1.f - pg) * p_copy[b];
    if (t < tlen[b]) {
      nll[b] -= logf(p + 1e-9f);
      cov[b] += covstep[b];
    }
  }
}

// ---------------------------------------------------------------------------
// Small helpers
// ---------------------------------------------------------------------------
__global__ void k_f2bf(const float* __restrict__ src, __bf16* __restrict__ dst, int n) {
  int i = blockIdx.x * 256 + threadIdx.x;
  if (i < n) dst[i] = (__bf16)src[i];
}

__global__ void k_init(float* __restrict__ coverage, float* __restrict__ nll, float* __restrict__ cov) {
  for (int i = threadIdx.x; i < 16 * TIN_; i += blockDim.x) coverage[i] = 0.f;
  if (threadIdx.x < 16) { nll[threadIdx.x] = 0.f; cov[threadIdx.x] = 0.f; }
}

__global__ void k_build_tokin(const int* __restrict__ tgt, int* __restrict__ tok) {
  for (int i = threadIdx.x; i < 16 * TTGT_; i += blockDim.x) {
    int b = i / TTGT_, t = i % TTGT_;
    int v;
    if (t == 0) v = START_;
    else { v = tgt[b * TTGT_ + t - 1]; if (v >= V_ || v < 0) v = UNK_; }
    tok[b * TTGT_ + t] = v;
  }
}

// ids [B,Tlen] -> bf16 out time-major [Tlen,B,128]
__global__ void k_embed(const int* __restrict__ ids, const float* __restrict__ emb,
                        __bf16* __restrict__ out, int Tlen) {
  int blk = blockIdx.x;             // blk = t*16 + b
  int t = blk >> 4, b = blk & 15;
  int id = ids[b * Tlen + t];
  if (id >= V_ || id < 0) id = UNK_;
  int d = threadIdx.x;              // 128 threads
  out[(size_t)blk * 128 + d] = (__bf16)emb[(size_t)id * 128 + d];
}

__global__ void k_output(const float* __restrict__ nll, const float* __restrict__ cov,
                         float* __restrict__ out) {
  if (threadIdx.x == 0) {
    float a = 0.f, c = 0.f;
    for (int b = 0; b < 16; ++b) { a += nll[b]; c += cov[b]; }
    out[0] = a; out[1] = c; out[2] = a + c;
  }
}

// ---------------------------------------------------------------------------
extern "C" void kernel_launch(void* const* d_in, const int* in_sizes, int n_in,
                              void* d_out, int out_size, void* d_ws, size_t ws_size,
                              hipStream_t stream) {
  (void)in_sizes; (void)n_in; (void)out_size; (void)ws_size;
  const int*   input_ids  = (const int*)d_in[0];
  const int*   target_ids = (const int*)d_in[1];
  /* input_lengths d_in[2] unused: full-length */
  const int*   tgt_len    = (const int*)d_in[3];
  const float* emb        = (const float*)d_in[4];
  const float* enc_Wih0   = (const float*)d_in[5];
  const float* enc_Whh0   = (const float*)d_in[6];
  const float* enc_b0     = (const float*)d_in[7];
  const float* enc_Wih1   = (const float*)d_in[8];
  const float* enc_Whh1   = (const float*)d_in[9];
  const float* enc_b1     = (const float*)d_in[10];
  const float* W_e2dh     = (const float*)d_in[11];
  const float* b_e2dh     = (const float*)d_in[12];
  const float* W_e2dc     = (const float*)d_in[13];
  const float* b_e2dc     = (const float*)d_in[14];
  const float* dec_Wih0   = (const float*)d_in[15];
  const float* dec_Whh0   = (const float*)d_in[16];
  const float* dec_b0     = (const float*)d_in[17];
  const float* dec_Wih1   = (const float*)d_in[18];
  const float* dec_Whh1   = (const float*)d_in[19];
  const float* dec_b1     = (const float*)d_in[20];
  const float* W_enc_attn = (const float*)d_in[21];
  const float* b_enc_attn = (const float*)d_in[22];
  const float* W_dec_attn = (const float*)d_in[23];
  const float* w_cov      = (const float*)d_in[24];
  const float* v_attn     = (const float*)d_in[25];
  const float* W_v1       = (const float*)d_in[26];
  const float* b_v1       = (const float*)d_in[27];
  const float* W_v2       = (const float*)d_in[28];
  const float* b_v2       = (const float*)d_in[29];
  const float* w_ctx_sw   = (const float*)d_in[30];
  const float* b_ctx_sw   = (const float*)d_in[31];
  const float* w_dec_sw   = (const float*)d_in[32];
  const float* w_emb_sw   = (const float*)d_in[33];

  float* ws = (float*)d_ws;
  size_t off = 0;
  auto alloc  = [&](size_t n)   { float* p = ws + off; off += n; return p; };
  auto allocb = [&](size_t nbf) { __bf16* p = (__bf16*)(ws + off); off += nbf / 2; return p; };

  __bf16* x0b      = allocb((size_t)TIN_ * 16 * 128);   // [T,16,128] bf16
  __bf16* hseq0b   = allocb((size_t)TIN_ * 16 * 512);   // [T,16,512] bf16
  float*  hseq1    = alloc((size_t)TIN_ * 16 * 512);    // enc_states f32
  float*  enc_attn = alloc((size_t)TIN_ * 16 * 256);
  float*  hTs      = alloc(2 * 16 * 512);
  float*  cTs      = alloc(2 * 16 * 512);
  float*  dh       = alloc(2 * 16 * 256);
  float*  dc       = alloc(2 * 16 * 256);
  __bf16* edecb    = allocb((size_t)TTGT_ * 16 * 128);
  int*    tok_in   = (int*)alloc(16 * TTGT_);
  float*  coverage = alloc(16 * TIN_);
  float*  scores   = alloc(16 * TIN_);
  float*  hc       = alloc(16 * 768);
  float*  dproj    = alloc(16 * 256);
  float*  hsw      = alloc(16);
  float*  p_copy   = alloc(16);
  float*  covstep  = alloc(16);
  float*  hv       = alloc(16 * 512);
  __bf16* hvb      = allocb(16 * 512);
  float*  logits   = alloc((size_t)16 * 32000);
  float*  mx       = alloc(16);
  float*  sm       = alloc(16);
  float*  nll      = alloc(16);
  float*  cov      = alloc(16);
  // bf16 weight copies
  __bf16* bWih0e = allocb((size_t)2 * 1024 * 128);
  __bf16* bWhh0e = allocb((size_t)2 * 1024 * 256);
  __bf16* bWih1e = allocb((size_t)2 * 1024 * 512);
  __bf16* bWhh1e = allocb((size_t)2 * 1024 * 256);
  __bf16* bWih0d = allocb((size_t)1024 * 128);
  __bf16* bWhh0d = allocb((size_t)1024 * 256);
  __bf16* bWih1d = allocb((size_t)1024 * 256);
  __bf16* bWhh1d = allocb((size_t)1024 * 256);
  __bf16* bWdatt = allocb((size_t)256 * 256);
  __bf16* bWenca = allocb((size_t)256 * 512);
  __bf16* bWe2dh = allocb((size_t)256 * 512);
  __bf16* bWe2dc = allocb((size_t)256 * 512);
  __bf16* bWv1   = allocb((size_t)512 * 768);
  __bf16* bWv2   = allocb((size_t)(V_ - END_) * 512);

  auto cvt = [&](const float* s, __bf16* d, size_t n) {
    k_f2bf<<<(unsigned)((n + 255) / 256), 256, 0, stream>>>(s, d, (int)n);
  };
  cvt(enc_Wih0, bWih0e, (size_t)2 * 1024 * 128);
  cvt(enc_Whh0, bWhh0e, (size_t)2 * 1024 * 256);
  cvt(enc_Wih1, bWih1e, (size_t)2 * 1024 * 512);
  cvt(enc_Whh1, bWhh1e, (size_t)2 * 1024 * 256);
  cvt(dec_Wih0, bWih0d, (size_t)1024 * 128);
  cvt(dec_Whh0, bWhh0d, (size_t)1024 * 256);
  cvt(dec_Wih1, bWih1d, (size_t)1024 * 256);
  cvt(dec_Whh1, bWhh1d, (size_t)1024 * 256);
  cvt(W_dec_attn, bWdatt, (size_t)256 * 256);
  cvt(W_enc_attn, bWenca, (size_t)256 * 512);
  cvt(W_e2dh, bWe2dh, (size_t)256 * 512);
  cvt(W_e2dc, bWe2dc, (size_t)256 * 512);
  cvt(W_v1, bWv1, (size_t)512 * 768);
  cvt(W_v2, bWv2, (size_t)(V_ - END_) * 512);

  // ---- setup ----
  k_init<<<1, 256, 0, stream>>>(coverage, nll, cov);
  k_build_tokin<<<1, 256, 0, stream>>>(target_ids, tok_in);
  k_embed<<<TIN_ * 16, 128, 0, stream>>>(input_ids, emb, x0b, TIN_);
  k_embed<<<TTGT_ * 16, 128, 0, stream>>>(tok_in, emb, edecb, TTGT_);

  // ---- encoder (2 bidirectional layers, persistent kernels) ----
  k_lstm<<<2, 512, 0, stream>>>(x0b, 128, bWih0e, bWhh0e, enc_b0,
                                hseq0b, (float*)nullptr, hTs, cTs, TIN_);
  k_lstm<<<2, 512, 0, stream>>>(hseq0b, 512, bWih1e, bWhh1e, enc_b1,
                                (__bf16*)nullptr, hseq1, hTs + 16 * 512, cTs + 16 * 512, TIN_);

  // enc_attn = enc_states @ W_enc_attn^T + b : [6400,512] x [256,512]^T
  k_gemm_fb<<<dim3(16, 400), 32, 0, stream>>>(hseq1, 512, bWenca, 512, b_enc_attn,
                                              enc_attn, (__bf16*)nullptr, 256, 512, 0);
  // decoder init state: [32,512] x [256,512]^T
  k_gemm_fb<<<dim3(16, 2), 32, 0, stream>>>(hTs, 512, bWe2dh, 512, b_e2dh,
                                            dh, (__bf16*)nullptr, 256, 512, 0);
  k_gemm_fb<<<dim3(16, 2), 32, 0, stream>>>(cTs, 512, bWe2dc, 512, b_e2dc,
                                            dc, (__bf16*)nullptr, 256, 512, 0);

  // ---- decoder loop ----
  for (int t = 0; t < TTGT_; ++t) {
    k_dec_step<<<1, 512, 0, stream>>>(t, edecb,
        bWih0d, bWhh0d, dec_b0, bWih1d, bWhh1d, dec_b1,
        bWdatt, w_dec_sw, w_emb_sw, b_ctx_sw,
        dh, dc, hc, dproj, hsw);
    k_scores<<<16 * TIN_, 32, 0, stream>>>(enc_attn, dproj, coverage, w_cov, v_attn,
                                           input_ids, scores);
    k_attn_ctx<<<16, 512, 0, stream>>>(t, scores, hseq1, input_ids, target_ids,
                                       coverage, hc, p_copy, covstep);
    // hv = relu([h1|ctx] @ W_v1^T + b_v1): also emit bf16 copy for logits A
    k_gemm_fb<<<dim3(32, 1), 32, 0, stream>>>(hc, 768, bWv1, 768, b_v1,
                                              hv, hvb, 512, 768, 1);
    // logits = hv @ W_v2^T + b_v2: pure-bf16 operands, 2 N-tiles per wave
    k_gemm_bb2<<<dim3(1000, 1), 32, 0, stream>>>(hvb, 512, bWv2, 512, b_v2,
                                                 logits, 32000, V_ - END_, 512);
    k_softmax_stats<<<16, 1024, 0, stream>>>(logits, mx, sm);
    k_finalize<<<16, 32, 0, stream>>>(t, logits, mx, sm, hc, w_ctx_sw, hsw,
                                      p_copy, covstep, target_ids, tgt_len, nll, cov);
  }

  k_output<<<1, 32, 0, stream>>>(nll, cov, (float*)d_out);
}
